// WindowAttention_78494822302232
// MI455X (gfx1250) — matrix operand order repaired and checked by reading, not compiled, not run on hardware
//
#include <hip/hip_runtime.h>
#include <hip/hip_bf16.h>

typedef __bf16 bf16;
typedef __attribute__((ext_vector_type(16))) __bf16 v16bf;
typedef __attribute__((ext_vector_type(8)))  float  v8f;

#define NTOK 64
#define DIM  256
#define NH   8
#define HD   32
#define NWIN 64
#define CPBH 512
#define TBLN 225            // (2*8-1)*(2*8-1)

// LDS row pitches, padded +8 bf16 (16B) to break 256B bank periodicity while
// keeping every 16B ds_load_b128 chunk aligned.  64 banks x 4B:
//  - pitch 528B = 132 dwords -> lane start banks {4i}  (conflict-free)
//  - pitch 144B =  36 dwords -> lane start banks {36i mod 64} (conflict-free)
#define ROWP  (DIM + 8)     // 264 bf16 = 528 B
#define VROWP (NTOK + 8)    // 72 bf16  = 144 B

__device__ __forceinline__ v8f wmma_bf16(v16bf a, v16bf b, v8f c) {
  // emits v_wmma_f32_16x16x32_bf16
  return __builtin_amdgcn_wmma_f32_16x16x32_bf16(
      /*neg_a=*/false, a, /*neg_b=*/false, b,
      /*c_mod=*/(short)0, c, /*reuse_a=*/false, /*reuse_b=*/false);
}

// -------- prep: bf16 weights, logit scale, fused qkv bias, CPB MLP table ---
__global__ void prep_kernel(const float* __restrict__ qkv_w,
                            const float* __restrict__ proj_w,
                            const float* __restrict__ logit_scale,
                            const float* __restrict__ cpb_w1,
                            const float* __restrict__ cpb_b1,
                            const float* __restrict__ cpb_w2,
                            const float* __restrict__ coords,
                            const float* __restrict__ q_bias,
                            const float* __restrict__ v_bias,
                            bf16* __restrict__ qkvw_b,
                            bf16* __restrict__ projw_b,
                            float* __restrict__ scale,
                            float* __restrict__ table,
                            float* __restrict__ qkvb)
{
  int idx = blockIdx.x * 256 + threadIdx.x;
  if (idx < 768 * 256) { qkvw_b[idx] = (bf16)qkv_w[idx]; return; }
  idx -= 768 * 256;
  if (idx < 256 * 256) { projw_b[idx] = (bf16)proj_w[idx]; return; }
  idx -= 256 * 256;
  if (idx < NH) {
    scale[idx] = __expf(fminf(logit_scale[idx], 4.605170185988091f)); // log(100)
    return;
  }
  idx -= NH;
  if (idx < 3 * DIM) {  // fused qkv bias: [q_bias, 0, v_bias]
    qkvb[idx] = (idx < DIM) ? q_bias[idx] : (idx < 2 * DIM ? 0.f : v_bias[idx - 2 * DIM]);
    return;
  }
  idx -= 3 * DIM;
  if (idx < TBLN * NH) {
    int t = idx / NH, h = idx % NH;
    float c0 = coords[t * 2 + 0], c1 = coords[t * 2 + 1];
    float acc = 0.f;
    for (int j = 0; j < CPBH; ++j) {
      float hv = fmaxf(0.f, c0 * cpb_w1[j * 2] + c1 * cpb_w1[j * 2 + 1] + cpb_b1[j]);
      acc += cpb_w2[h * CPBH + j] * hv;
    }
    table[t * NH + h] = acc;
  }
}

// -------- gather relative-position bias: bias[h][n][m] = 16*sigmoid(.) -----
__global__ void bias_kernel(const float* __restrict__ table,
                            const int* __restrict__ rel_idx,
                            float* __restrict__ bias)
{
  int idx = blockIdx.x * 256 + threadIdx.x;
  if (idx >= NH * NTOK * NTOK) return;
  int h = idx >> 12;          // /4096
  int r = idx & 4095;         // n*64+m
  int t = rel_idx[r];
  float v = table[t * NH + h];
  bias[idx] = 16.f / (1.f + __expf(-v));
}

// -------- fused window attention: one workgroup (8 waves) per window -------
// LDS map (bytes), 142336 total -> 2 workgroups/WGP:
//   sX    [64][264] bf16      0 .. 33792   (P1->P2)   } aliased by sScr (P3/P4)
//   sQ    [64][264] bf16  33792 .. 67584   (P2->P3a)  } aliased by sOut (P4->P5)
//   sK    [64][264] bf16  67584 .. 101376  (P2->P3a)
//   sVt  [256][72]  bf16 101376 .. 138240  (P2->P4)   v transposed
//   sSq   [64][8]   f32  138240 .. 140288  1/||q||
//   sSk   [64][8]   f32  140288 .. 142336  1/||k||
__global__ __launch_bounds__(256)
void win_attn_kernel(const float* __restrict__ x,
                     const float* __restrict__ mask,
                     const float* __restrict__ qkvb,
                     const float* __restrict__ proj_b,
                     const bf16* __restrict__ qkvw,
                     const bf16* __restrict__ projw,
                     const float* __restrict__ scale,
                     const float* __restrict__ bias,
                     float* __restrict__ out)
{
  extern __shared__ char smem[];
  bf16 (* __restrict__ sX)[ROWP]   = (bf16(*)[ROWP])(smem);
  bf16 (* __restrict__ sQ)[ROWP]   = (bf16(*)[ROWP])(smem + 33792);
  bf16 (* __restrict__ sK)[ROWP]   = (bf16(*)[ROWP])(smem + 67584);
  bf16 (* __restrict__ sVt)[VROWP] = (bf16(*)[VROWP])(smem + 101376);
  float* sSq = (float*)(smem + 138240);
  float* sSk = (float*)(smem + 140288);
  bf16 (* __restrict__ sOut)[ROWP] = (bf16(*)[ROWP])(smem + 33792);  // over sQ

  const int tid  = threadIdx.x;
  const int wave = tid >> 5;
  const int lane = tid & 31;
  const int l16  = lane & 15;
  const int g    = lane >> 4;
  const int b    = blockIdx.x;

  // 16x64 macro-tile GEMM: one A-fragment shared by 4 column WMMAs per K-step.
  // A: LDS row-major [.][ROWP]; W: global bf16 [cols][DIM] accessed transposed.
  auto gemm4 = [&](const bf16 (*A)[ROWP], int arow, const bf16* __restrict__ W,
                   int m0, v8f acc[4]) {
#pragma unroll
    for (int kk = 0; kk < 8; ++kk) {
      int k0 = kk * 32;
      v16bf a;
#pragma unroll
      for (int e = 0; e < 16; ++e)
          a[e] = A[arow][k0 + e + 16 * g];
#pragma unroll
      for (int c = 0; c < 4; ++c) {
        v16bf bb;
#pragma unroll
        for (int e = 0; e < 16; ++e)
          bb[e] = W[(m0 + c * 16 + l16) * DIM + k0 + e + 16 * g];
        acc[c] = wmma_bf16(a, bb, acc[c]);
      }
    }
  };

  // ---- P1: stage x (f32 -> bf16), zero norm accumulators ----
  const float* xb = x + (size_t)b * NTOK * DIM;
  for (int i = tid; i < NTOK * DIM; i += 256)
    sX[i >> 8][i & 255] = (bf16)xb[i];
  for (int i = tid; i < 2 * NTOK * NH; i += 256) sSq[i] = 0.f; // sSq+sSk contiguous
  __syncthreads();

  // ---- P2: qkv = x @ qkv_w^T + [q_bias,0,v_bias] ----
  // Three branch-free sections (q / k / v), 16 macro-tiles each (4 row x 4 col).
  for (int tt = wave; tt < 16; tt += 8) {          // Q section
    int n0 = (tt & 3) * 16, m0 = (tt >> 2) * 64;
    v8f acc[4] = {{}, {}, {}, {}};
    gemm4(sX, n0 + l16, qkvw, m0, acc);
#pragma unroll
    for (int c = 0; c < 4; ++c) {
      int col = m0 + c * 16 + l16;
      float bm = qkvb[col];
#pragma unroll
      for (int v = 0; v < 8; ++v) {
        int r = n0 + v + 8 * g;
        float val = acc[c][v] + bm;
        sQ[r][col] = (bf16)val;
        atomicAdd(&sSq[r * NH + (col >> 5)], val * val);
      }
    }
  }
  for (int tt = wave; tt < 16; tt += 8) {          // K section
    int n0 = (tt & 3) * 16, m0 = (tt >> 2) * 64;
    v8f acc[4] = {{}, {}, {}, {}};
    gemm4(sX, n0 + l16, qkvw, 256 + m0, acc);
#pragma unroll
    for (int c = 0; c < 4; ++c) {
      int col = m0 + c * 16 + l16;                 // 0..255 within k block
      float bm = qkvb[256 + col];
#pragma unroll
      for (int v = 0; v < 8; ++v) {
        int r = n0 + v + 8 * g;
        float val = acc[c][v] + bm;
        sK[r][col] = (bf16)val;
        atomicAdd(&sSk[r * NH + (col >> 5)], val * val);
      }
    }
  }
  for (int tt = wave; tt < 16; tt += 8) {          // V section (store transposed)
    int n0 = (tt & 3) * 16, m0 = (tt >> 2) * 64;
    v8f acc[4] = {{}, {}, {}, {}};
    gemm4(sX, n0 + l16, qkvw, 512 + m0, acc);
#pragma unroll
    for (int c = 0; c < 4; ++c) {
      int col = m0 + c * 16 + l16;                 // col = h*32 + d
      float bm = qkvb[512 + col];
#pragma unroll
      for (int v = 0; v < 8; ++v) {
        int r = n0 + v + 8 * g;
        sVt[col][r] = (bf16)(acc[c][v] + bm);
      }
    }
  }
  __syncthreads();

  // ---- P2.5: sums -> inverse norms (F.normalize eps semantics) ----
  for (int i = tid; i < 2 * NTOK * NH; i += 256)
    sSq[i] = 1.f / fmaxf(sqrtf(sSq[i]), 1e-12f);
  __syncthreads();

  // ---- P3a: one wave per head; S = q_h @ k_h^T (K=32, 16 tiles in regs) ----
  const int h = wave;
  v8f S[4][4];
#pragma unroll
  for (int ti = 0; ti < 4; ++ti) {
    v16bf a;                                   // q fragment shared across tj
    int qrow = ti * 16 + l16;
#pragma unroll
    for (int e = 0; e < 16; ++e)
      a[e] = sQ[qrow][h * HD + e + 16 * g];
#pragma unroll
    for (int tj = 0; tj < 4; ++tj) {
      v16bf bb;
      int krow = tj * 16 + l16;
#pragma unroll
      for (int e = 0; e < 16; ++e)
        bb[e] = sK[krow][h * HD + e + 16 * g];
      v8f z = {};
      S[ti][tj] = wmma_bf16(a, bb, z);
    }
  }
  __syncthreads();  // sQ/sK reads done; their regions may now be reused

  // ---- P3b+P4 fused per 16-row chunk: scale+bias+mask -> softmax ->
  //      per-wave bf16 scratch (aliases sX) -> attn@v -> sOut (aliases sQ) --
  {
    float sc = scale[h];
    const float* biash = bias + (size_t)h * NTOK * NTOK;
    const float* maskw = mask + (size_t)(b & (NWIN - 1)) * NTOK * NTOK;
    float rsk[4];
#pragma unroll
    for (int tj = 0; tj < 4; ++tj) rsk[tj] = sSk[(tj * 16 + l16) * NH + h];
    // wave-private attn chunk [16][72] bf16 (2304 B each, 8 waves = 18 KB)
    bf16 (* __restrict__ scr)[VROWP] =
        (bf16(*)[VROWP])(smem + (size_t)h * 16 * VROWP * 2);

#pragma unroll
    for (int ti = 0; ti < 4; ++ti) {
#pragma unroll
      for (int v = 0; v < 8; ++v) {
        int rloc = v + 8 * g;
        int r = ti * 16 + rloc;
        float rq = sSq[r * NH + h];
        float vv[4];
        float mx = -1e30f;
#pragma unroll
        for (int tj = 0; tj < 4; ++tj) {
          int m = tj * 16 + l16;
          float val = S[ti][tj][v] * rq * rsk[tj] * sc
                    + biash[r * NTOK + m] + maskw[r * NTOK + m];
          vv[tj] = val;
          mx = fmaxf(mx, val);
        }
#pragma unroll
        for (int s = 1; s < 16; s <<= 1) mx = fmaxf(mx, __shfl_xor(mx, s, 32));
        float sum = 0.f;
#pragma unroll
        for (int tj = 0; tj < 4; ++tj) { vv[tj] = __expf(vv[tj] - mx); sum += vv[tj]; }
#pragma unroll
        for (int s = 1; s < 16; s <<= 1) sum += __shfl_xor(sum, s, 32);
        float inv = 1.f / sum;
#pragma unroll
        for (int tj = 0; tj < 4; ++tj)
          scr[rloc][tj * 16 + l16] = (bf16)(vv[tj] * inv);
      }
      __builtin_amdgcn_wave_barrier();   // order scr stores before scr loads

      // out[ti] = attn_chunk(16x64) @ v_h(64x32): attn fragment shared per K
      v8f acc[2] = {{}, {}};
#pragma unroll
      for (int kk = 0; kk < 2; ++kk) {
        int k0 = kk * 32;
        v16bf a;
#pragma unroll
        for (int e = 0; e < 16; ++e)
          a[e] = scr[l16][k0 + e + 16 * g];
#pragma unroll
        for (int dj = 0; dj < 2; ++dj) {
          v16bf bb;
          int brow = h * HD + dj * 16 + l16;   // row of v^T = (h*32+d)
#pragma unroll
          for (int e = 0; e < 16; ++e)
            bb[e] = sVt[brow][k0 + e + 16 * g];
          acc[dj] = wmma_bf16(a, bb, acc[dj]);
        }
      }
#pragma unroll
      for (int dj = 0; dj < 2; ++dj)
#pragma unroll
        for (int v = 0; v < 8; ++v)
          sOut[ti * 16 + v + 8 * g][h * HD + dj * 16 + l16] = (bf16)acc[dj][v];
      __builtin_amdgcn_wave_barrier();   // before next ti overwrites scr
    }
  }
  __syncthreads();

  // ---- P5: y = out @ proj_w^T + proj_b -> global f32 (16 macro-tiles) ----
  float* outb = out + (size_t)b * NTOK * DIM;
  for (int tt = wave; tt < 16; tt += 8) {
    int n0 = (tt & 3) * 16, m0 = (tt >> 2) * 64;
    v8f acc[4] = {{}, {}, {}, {}};
    gemm4(sOut, n0 + l16, projw, m0, acc);
#pragma unroll
    for (int c = 0; c < 4; ++c) {
      int col = m0 + c * 16 + l16;
      float pb = proj_b[col];
#pragma unroll
      for (int v = 0; v < 8; ++v) {
        int r = n0 + v + 8 * g;
        outb[r * DIM + col] = acc[c][v] + pb;
      }
    }
  }
}

extern "C" void kernel_launch(void* const* d_in, const int* in_sizes, int n_in,
                              void* d_out, int out_size, void* d_ws, size_t ws_size,
                              hipStream_t stream)
{
  (void)in_sizes; (void)n_in; (void)out_size; (void)ws_size;
  const float* x       = (const float*)d_in[0];
  const float* mask    = (const float*)d_in[1];
  const float* qkv_w   = (const float*)d_in[2];
  const float* q_bias  = (const float*)d_in[3];
  const float* v_bias  = (const float*)d_in[4];
  const float* logit_s = (const float*)d_in[5];
  const float* cpb_w1  = (const float*)d_in[6];
  const float* cpb_b1  = (const float*)d_in[7];
  const float* cpb_w2  = (const float*)d_in[8];
  const float* proj_w  = (const float*)d_in[9];
  const float* proj_b  = (const float*)d_in[10];
  const float* coords  = (const float*)d_in[11];
  const int*   rel_idx = (const int*)d_in[12];

  char* ws = (char*)d_ws;
  bf16*  qkvw_b  = (bf16*)ws;                              // 768*256 bf16
  bf16*  projw_b = (bf16*)(ws + 768 * 256 * 2);            // 256*256 bf16
  float* fws   = (float*)(ws + 768 * 256 * 2 + 256 * 256 * 2);
  float* scale = fws;          // 8
  float* table = fws + 8;      // 225*8 = 1800
  float* qkvb  = fws + 1808;   // 768
  float* bias  = fws + 4096;   // 8*64*64 = 32768

  int prep_total = 768 * 256 + 256 * 256 + NH + 3 * DIM + TBLN * NH;
  prep_kernel<<<(prep_total + 255) / 256, 256, 0, stream>>>(
      qkv_w, proj_w, logit_s, cpb_w1, cpb_b1, cpb_w2, coords, q_bias, v_bias,
      qkvw_b, projw_b, scale, table, qkvb);
  bias_kernel<<<(NH * NTOK * NTOK + 255) / 256, 256, 0, stream>>>(table, rel_idx, bias);

  const size_t smem = 142336;  // 139 KB aliased LDS -> 2 workgroups/WGP
  hipFuncSetAttribute((const void*)win_attn_kernel,
                      hipFuncAttributeMaxDynamicSharedMemorySize, (int)smem);
  win_attn_kernel<<<2048, 256, smem, stream>>>(
      x, mask, qkvb, proj_b, qkvw_b, projw_b, scale, bias, (float*)d_out);
}
